// CascadeHadamardSmoothLinear_49787260895359
// MI455X (gfx1250) — compile-verified, matrix operand order checked by reading, not run on hardware
//
#include <hip/hip_runtime.h>
#include <hip/hip_bf16.h>
#include <stdint.h>

typedef _Float16 half_t;
typedef _Float16 v16h __attribute__((ext_vector_type(16)));
typedef float    v8f  __attribute__((ext_vector_type(8)));
typedef unsigned int u32x4 __attribute__((ext_vector_type(4)));
typedef int          i32x4 __attribute__((ext_vector_type(4)));
typedef int          i32x8 __attribute__((ext_vector_type(8)));

#define GEMM_TM 128
#define GEMM_TN 128
#define GEMM_TK 64
#define LDS_STRIDE 72   // halves per LDS row (64 data + 8 pad) = 144B -> conflict-free b128 reads
#define BUF_HALVES (GEMM_TM * LDS_STRIDE)
#define BUF_BYTES  (BUF_HALVES * 2)

#if defined(__has_builtin)
#if __has_builtin(__builtin_amdgcn_tensor_load_to_lds) && __has_builtin(__builtin_amdgcn_s_wait_tensorcnt)
#define HAVE_TDM 1
#endif
#endif

// ---------------------------------------------------------------------------
// Kernel 1: convert weights f32 -> f16
// ---------------------------------------------------------------------------
__global__ __launch_bounds__(256)
void wconv_kernel(const float* __restrict__ w, half_t* __restrict__ wq, int n)
{
    int i = (blockIdx.x * 256 + threadIdx.x) * 8;
    if (i >= n) return;
    float4 a = *(const float4*)(w + i);
    float4 b = *(const float4*)(w + i + 4);
    union { half_t h[8]; uint4 q; } u;
    u.h[0] = (half_t)a.x; u.h[1] = (half_t)a.y; u.h[2] = (half_t)a.z; u.h[3] = (half_t)a.w;
    u.h[4] = (half_t)b.x; u.h[5] = (half_t)b.y; u.h[6] = (half_t)b.z; u.h[7] = (half_t)b.w;
    *(uint4*)(wq + i) = u.q;
}

// ---------------------------------------------------------------------------
// Kernel 2: fused smooth*sign -> FWHT-128 -> NVFP4 fake-quant -> f16
// ---------------------------------------------------------------------------
__global__ __launch_bounds__(256)
void act_kernel(const float* __restrict__ x, const float* __restrict__ ss,
                const float* __restrict__ sgn, half_t* __restrict__ aq, int K)
{
    const int m = blockIdx.x;
    const int t = threadIdx.x;
    const int kbase = t * 16;
    const float* xp = x + (size_t)m * K + kbase;

    float e[16];
    for (int j = 0; j < 4; ++j) {
        float4 xv = *(const float4*)(xp + j * 4);
        float4 sv = *(const float4*)(ss + kbase + j * 4);
        float4 gv = *(const float4*)(sgn + kbase + j * 4);
        e[j * 4 + 0] = xv.x * sv.x * gv.x;
        e[j * 4 + 1] = xv.y * sv.y * gv.y;
        e[j * 4 + 2] = xv.z * sv.z * gv.z;
        e[j * 4 + 3] = xv.w * sv.w * gv.w;
    }

    for (int len = 1; len < 16; len <<= 1)
        for (int i = 0; i < 16; ++i)
            if (!(i & len)) {
                float a = e[i], b = e[i | len];
                e[i] = a + b;
                e[i | len] = a - b;
            }
    for (int s = 0; s < 3; ++s) {
        float sgnv = ((t >> s) & 1) ? -1.0f : 1.0f;
        for (int i = 0; i < 16; ++i) {
            float o = __shfl_xor(e[i], 1 << s, 32);
            e[i] = o + sgnv * e[i];
        }
    }
    const float rs = 0.08838834764831845f; // 1/sqrt(128)
    for (int i = 0; i < 16; ++i) e[i] *= rs;

    float amax = 0.0f;
    for (int i = 0; i < 16; ++i) amax = fmaxf(amax, fabsf(e[i]));
    amax = fmaxf(amax, 1e-12f);
    const float scale = amax * (1.0f / 6.0f);
    const float inv = 6.0f / amax;

    union { half_t h[16]; uint4 q[2]; } o;
    for (int i = 0; i < 16; ++i) {
        float v = fabsf(e[i]) * inv;
        float lev = (v <= 0.25f) ? 0.0f :
                    (v <= 0.75f) ? 0.5f :
                    (v <= 1.25f) ? 1.0f :
                    (v <= 1.75f) ? 1.5f :
                    (v <= 2.5f)  ? 2.0f :
                    (v <= 3.5f)  ? 3.0f :
                    (v <= 5.0f)  ? 4.0f : 6.0f;
        o.h[i] = (half_t)copysignf(lev * scale, e[i]);
    }
    half_t* dst = aq + (size_t)m * K + kbase;
    *(uint4*)(dst)     = o.q[0];
    *(uint4*)(dst + 8) = o.q[1];
}

// ---------------------------------------------------------------------------
// Kernel 3: WMMA GEMM with TDM-staged LDS double buffering
// ---------------------------------------------------------------------------
#ifdef HAVE_TDM
// DMA one 128-row x 64-half tile (row stride K halves) into LDS at lds_off,
// writing LDS rows padded to 72 halves via TDM pad (every 32 DWORDs insert 4).
__device__ __forceinline__ void tdm_load_tile(const half_t* gsrc, unsigned int lds_off, int K)
{
    unsigned long long ga = (unsigned long long)(uintptr_t)gsrc;
    u32x4 g0;
    g0[0] = 1u;                                      // count=1 valid descriptor
    g0[1] = lds_off;                                 // LDS byte address
    g0[2] = (unsigned int)ga;                        // global_addr[31:0]
    g0[3] = (unsigned int)(ga >> 32) | (2u << 30);   // global_addr[56:32] | type=2

    i32x8 g1;
    g1[0] = (1 << 16)      // data_size = 2 bytes
          | (1 << 20)      // pad_enable
          | (4 << 22)      // pad_interval: every 32 DWORDs (= 64 halves)
          | (3 << 25);     // pad_amount: 4 DWORDs (= 8 halves)
    g1[1] = (int)(((unsigned int)K & 0xFFFFu) << 16);                  // tensor_dim0[15:0]
    g1[2] = (int)((((unsigned int)K >> 16) & 0xFFFFu) | (128u << 16)); // dim0 hi | tensor_dim1=128
    g1[3] = (int)(64u << 16);                                          // tensor_dim1 hi=0 | tile_dim0=64
    g1[4] = 128;                                                       // tile_dim1=128, tile_dim2=0
    g1[5] = K;                                                         // tensor_dim0_stride[31:0]
    g1[6] = 0;
    g1[7] = 0;

    i32x4 gz4 = (i32x4)0;
    i32x8 gz8 = (i32x8)0;
    __builtin_amdgcn_tensor_load_to_lds(g0, g1, gz4, gz4, gz8, 0);
}
#endif

__global__ __launch_bounds__(256, 1)
void gemm_kernel(const half_t* __restrict__ A, const half_t* __restrict__ W,
                 const float* __restrict__ bias, float* __restrict__ out,
                 int M, int N, int K)
{
    __shared__ __align__(16) half_t As[2][BUF_HALVES];
    __shared__ __align__(16) half_t Bs[2][BUF_HALVES];

    const int tid  = threadIdx.x;
    const int lane = tid & 31;
    const int wave = tid >> 5;
    const int rowM = blockIdx.y * GEMM_TM;
    const int colN = blockIdx.x * GEMM_TN;
    const int wm = (wave & 1) * 64;
    const int wn = (wave >> 1) * 32;
    const int lrow = lane & 15;
    const int hsel = lane >> 4;

    v8f acc[4][2];
    for (int nt = 0; nt < 2; ++nt) {
        float bv = bias[colN + wn + nt * 16 + lrow];
        for (int mt = 0; mt < 4; ++mt)
            for (int e = 0; e < 8; ++e) acc[mt][nt][e] = bv;
    }

    const int NKB = K / GEMM_TK;
    const half_t* Abase = A + (size_t)rowM * K;
    const half_t* Wbase = W + (size_t)colN * K;

#ifdef HAVE_TDM
    const unsigned int asOff = (unsigned int)(uintptr_t)&As[0][0];
    const unsigned int bsOff = (unsigned int)(uintptr_t)&Bs[0][0];
    // prologue: stage 0 via TDM
    if (wave == 0) { tdm_load_tile(Abase, asOff, K);
                     __builtin_amdgcn_s_wait_tensorcnt(0); }
    if (wave == 1) { tdm_load_tile(Wbase, bsOff, K);
                     __builtin_amdgcn_s_wait_tensorcnt(0); }
    __syncthreads();
#else
    {
        const int r0 = tid >> 3;
        const int c  = (tid & 7) * 8;
        for (int j = 0; j < 4; ++j) {
            int rr = r0 + j * 32;
            *(uint4*)(&As[0][rr * LDS_STRIDE + c]) = *(const uint4*)(Abase + (size_t)rr * K + c);
            *(uint4*)(&Bs[0][rr * LDS_STRIDE + c]) = *(const uint4*)(Wbase + (size_t)rr * K + c);
        }
    }
    __syncthreads();
#endif

    for (int kb = 0; kb < NKB; ++kb) {
        const int cur = kb & 1;
        const bool more = (kb + 1) < NKB;
        const int nb = cur ^ 1;

#ifdef HAVE_TDM
        if (more) {
            const size_t koff = (size_t)(kb + 1) * GEMM_TK;
            if (wave == 0) tdm_load_tile(Abase + koff, asOff + (unsigned)nb * BUF_BYTES, K);
            if (wave == 1) tdm_load_tile(Wbase + koff, bsOff + (unsigned)nb * BUF_BYTES, K);
        }
#else
        if (more) {
            const int r0 = tid >> 3;
            const int c  = (tid & 7) * 8;
            const size_t koff = (size_t)(kb + 1) * GEMM_TK + c;
            for (int j = 0; j < 4; ++j) {
                int rr = r0 + j * 32;
                *(uint4*)(&As[nb][rr * LDS_STRIDE + c]) = *(const uint4*)(Abase + (size_t)rr * K + koff);
                *(uint4*)(&Bs[nb][rr * LDS_STRIDE + c]) = *(const uint4*)(Wbase + (size_t)rr * K + koff);
            }
        }
#endif

        for (int kk = 0; kk < GEMM_TK; kk += 32) {
            v16h af[4], bf[2];
            for (int mt = 0; mt < 4; ++mt) {
                const half_t* p = &As[cur][(wm + mt * 16 + lrow) * LDS_STRIDE + kk + hsel * 8];
                union { v16h v; uint4 q[2]; } u;
                u.q[0] = *(const uint4*)p;
                u.q[1] = *(const uint4*)(p + 16);
                af[mt] = u.v;
            }
            for (int nt = 0; nt < 2; ++nt) {
                const half_t* p = &Bs[cur][(wn + nt * 16 + lrow) * LDS_STRIDE + kk + hsel * 8];
                union { v16h v; uint4 q[2]; } u;
                u.q[0] = *(const uint4*)p;
                u.q[1] = *(const uint4*)(p + 16);
                bf[nt] = u.v;
            }
            for (int mt = 0; mt < 4; ++mt)
                for (int nt = 0; nt < 2; ++nt)
                    acc[mt][nt] = __builtin_amdgcn_wmma_f32_16x16x32_f16(
                        false, af[mt], false, bf[nt], (short)0, acc[mt][nt], false, false);
        }

#ifdef HAVE_TDM
        if (more && wave < 2) __builtin_amdgcn_s_wait_tensorcnt(0);
#endif
        __syncthreads();
    }

    for (int mt = 0; mt < 4; ++mt)
        for (int nt = 0; nt < 2; ++nt) {
            const int col = colN + wn + nt * 16 + lrow;
            for (int e = 0; e < 8; ++e) {
                const int row = rowM + wm + mt * 16 + e + hsel * 8;
                out[(size_t)row * N + col] = acc[mt][nt][e];
            }
        }
}

// ---------------------------------------------------------------------------
extern "C" void kernel_launch(void* const* d_in, const int* in_sizes, int n_in,
                              void* d_out, int out_size, void* d_ws, size_t ws_size,
                              hipStream_t stream)
{
    const float* x    = (const float*)d_in[0];
    const float* ss   = (const float*)d_in[1];
    const float* sgn  = (const float*)d_in[2];
    // d_in[3] (H_block) unused: FWHT computes the identical Sylvester-Hadamard product
    const float* w    = (const float*)d_in[4];
    const float* bias = (const float*)d_in[5];
    float* out = (float*)d_out;

    const int K = in_sizes[1];              // Din
    const int M = in_sizes[0] / K;          // B*S
    const int N = in_sizes[4] / K;          // Dout

    half_t* Aq = (half_t*)d_ws;             // M*K halves
    half_t* Wq = Aq + (size_t)M * K;        // N*K halves

    {
        int n = N * K;
        int blocks = (n / 8 + 255) / 256;
        wconv_kernel<<<blocks, 256, 0, stream>>>(w, Wq, n);
    }
    act_kernel<<<M, 256, 0, stream>>>(x, ss, sgn, Aq, K);

    dim3 grid(N / GEMM_TN, M / GEMM_TM);
    gemm_kernel<<<grid, 256, 0, stream>>>(Aq, Wq, bias, out, M, N, K);
}
